// RecursiveTransformerBlock_29180007809110
// MI455X (gfx1250) — matrix operand order repaired and checked by reading, not compiled, not run on hardware
//
#include <hip/hip_runtime.h>

// ---------------- types & helpers ----------------
typedef __attribute__((ext_vector_type(16))) __bf16 v16bf;
typedef __attribute__((ext_vector_type(8)))  float  v8f;

constexpr int NB = 2;       // batch
constexpr int NS = 2048;    // seq len
constexpr int ND = 1024;    // model dim
constexpr int NI = 4096;    // mlp dim
constexpr int NH = 16;      // heads
constexpr int HD = 64;      // head dim
constexpr int NT = NB * NS; // tokens = 4096
constexpr int WIN = 256;    // sliding window

__device__ __forceinline__ __bf16 f2bf(float f) {
  unsigned u = __float_as_uint(f);
  u += 0x7fffu + ((u >> 16) & 1u);          // round-to-nearest-even
  unsigned short h = (unsigned short)(u >> 16);
  return __builtin_bit_cast(__bf16, h);
}
__device__ __forceinline__ float bf2f(__bf16 b) {
  unsigned short h = __builtin_bit_cast(unsigned short, b);
  return __uint_as_float(((unsigned)h) << 16);
}
__device__ __forceinline__ unsigned lds_addr(const void* p) {
  // as3 -> generic cast keeps LDS byte address in the low 32 bits
  return (unsigned)(unsigned long long)(uintptr_t)p;
}

// ---------------- elementwise kernels ----------------
__global__ __launch_bounds__(256) void cvt_f32_bf16(const float* __restrict__ src,
                                                    __bf16* __restrict__ dst, size_t n) {
  size_t i = (size_t)blockIdx.x * blockDim.x + threadIdx.x;
  size_t st = (size_t)gridDim.x * blockDim.x;
  for (; i < n; i += st) dst[i] = f2bf(src[i]);
}

__global__ __launch_bounds__(256) void silu_mul_kernel(__bf16* __restrict__ a,
                                                       const __bf16* __restrict__ g3, size_t n) {
  size_t i = (size_t)blockIdx.x * blockDim.x + threadIdx.x;
  size_t st = (size_t)gridDim.x * blockDim.x;
  for (; i < n; i += st) {
    float x = bf2f(a[i]);
    float y = x / (1.f + __expf(-x));       // silu
    a[i] = f2bf(y * bf2f(g3[i]));
  }
}

// one block per token row, D=1024
__global__ __launch_bounds__(256) void ln_kernel(const float* __restrict__ x,
                                                 const float* __restrict__ g,
                                                 const float* __restrict__ b,
                                                 __bf16* __restrict__ out) {
  __shared__ float sh[256], sh2[256];
  const int row = blockIdx.x, tid = threadIdx.x;
  const float* xr = x + (size_t)row * ND;
  float s = 0.f, s2 = 0.f;
  for (int i = tid; i < ND; i += 256) { float v = xr[i]; s += v; s2 += v * v; }
  sh[tid] = s; sh2[tid] = s2; __syncthreads();
  for (int st = 128; st > 0; st >>= 1) {
    if (tid < st) { sh[tid] += sh[tid + st]; sh2[tid] += sh2[tid + st]; }
    __syncthreads();
  }
  const float mean = sh[0] / ND;
  const float var  = sh2[0] / ND - mean * mean;
  const float r    = rsqrtf(var + 1e-6f);
  for (int i = tid; i < ND; i += 256)
    out[(size_t)row * ND + i] = f2bf((xr[i] - mean) * r * g[i] + b[i]);
}

// rotate-half RoPE in place on bf16 [T, D] (head-major in D); scale folds 1/sqrt(hd) for Q
__global__ __launch_bounds__(256) void rope_kernel(__bf16* __restrict__ x, float scale) {
  const int p = blockIdx.x * 256 + threadIdx.x;      // pair index, total NT*512
  if (p >= NT * (ND / 2)) return;
  const int t = p >> 9;            // token
  const int q = p & 511;
  const int h = q >> 5;            // head
  const int d = q & 31;            // rotary pair index
  const int pos = t & (NS - 1);    // position within sequence (NS power of 2)
  const float inv = __expf(-(float)d * 0.28782313662425572f); // ln(10000)/32
  float c, s; __sincosf((float)pos * inv, &c, &s);
  const size_t base = (size_t)t * ND + h * HD + d;
  const float x1 = bf2f(x[base]), x2 = bf2f(x[base + 32]);
  x[base]      = f2bf((x1 * c - x2 * s) * scale);
  x[base + 32] = f2bf((x2 * c + x1 * s) * scale);
}

// ---- bf16 WMMA GEMM: C[MxN] = A[MxK] @ B[KxN] (+resid), double-buffered async LDS pipe ----
constexpr int BM = 128, BN = 64, BK = 32, PAD = 8;

template <bool F32OUT, bool HASRES>
__global__ __launch_bounds__(256) void gemm_bf16(const __bf16* __restrict__ A,
                                                 const __bf16* __restrict__ Bm,
                                                 int M, int N, int K,
                                                 float* __restrict__ outF,
                                                 __bf16* __restrict__ outB,
                                                 const float* __restrict__ resid) {
  __shared__ __bf16 As[2][BM][BK + PAD];
  __shared__ __bf16 Bs[2][BN][BK + PAD];   // transposed: Bs[buf][n][k]
  const int m0 = blockIdx.y * BM, n0 = blockIdx.x * BN;
  const int tid = threadIdx.x;
  const int lane = tid & 31, wv = tid >> 5;   // 8 waves
  const int half = lane >> 4, l15 = lane & 15;

  // staging coordinates (fixed per thread)
  const int sa_row = tid >> 1, sa_c0 = (tid & 1) * 16;   // A: 32B contiguous per thread
  const int sb_kr  = tid >> 3, sb_c0 = (tid & 7) * 8;    // B: 8 bf16, transpose-scatter

  // stage tile k0 into buffer buf: A via async DMA to LDS, B via VGPR transpose
  auto stage = [&](int k0, int buf) {
    const __bf16* srcA = A + (size_t)(m0 + sa_row) * K + k0 + sa_c0;
    const unsigned lds = lds_addr(&As[buf][sa_row][sa_c0]);
    const unsigned long long ga = (unsigned long long)(uintptr_t)srcA;
    asm volatile("global_load_async_to_lds_b128 %0, %1, off"
                 :: "v"(lds), "v"(ga) : "memory");
    asm volatile("global_load_async_to_lds_b128 %0, %1, off offset:16"
                 :: "v"(lds), "v"(ga) : "memory");
    const __bf16* srcB = Bm + (size_t)(k0 + sb_kr) * N + n0 + sb_c0;
#pragma unroll
    for (int e = 0; e < 8; ++e) Bs[buf][sb_c0 + e][sb_kr] = srcB[e];
  };

  v8f acc[4];
  const v8f zero8 = {0.f,0.f,0.f,0.f,0.f,0.f,0.f,0.f};
#pragma unroll
  for (int t = 0; t < 4; ++t) acc[t] = zero8;

  // prologue: fill buffer 0
  stage(0, 0);
  asm volatile("s_wait_asynccnt 0x0" ::: "memory");
  __syncthreads();

  for (int k0 = 0; k0 < K; k0 += BK) {
    const int buf = (k0 / BK) & 1;
    const bool more = (k0 + BK) < K;
    if (more) stage(k0 + BK, buf ^ 1);   // overlap next-tile DMA with compute

    // A fragment: 16x32 per wave (rows wv*16..wv*16+15)
    const int arow = wv * 16 + l15;
    const int koff = half * 8;
    v16bf af;
#pragma unroll
    for (int e = 0; e < 8; ++e) {
      af[e]     = As[buf][arow][koff + e];        // K = 0..7 / 8..15
      af[8 + e] = As[buf][arow][16 + koff + e];   // K = 16..23 / 24..31
    }
#pragma unroll
    for (int t = 0; t < 4; ++t) {
      const int bc = t * 16 + l15;
      const int kh = half * 8;
      v16bf bfv;
#pragma unroll
      for (int e = 0; e < 16; ++e) bfv[e] = Bs[buf][bc][kh + e + (e & 8)];
      acc[t] = __builtin_amdgcn_wmma_f32_16x16x32_bf16(false, af, false, bfv,
                                                       (short)0, acc[t], false, false);
    }
    if (more) asm volatile("s_wait_asynccnt 0x0" ::: "memory");
    __syncthreads();
  }

  // epilogue (C layout: VGPR r -> row r + 8*half, col = lane%16)
  const int rbase = m0 + wv * 16 + half * 8;
#pragma unroll
  for (int t = 0; t < 4; ++t) {
    const int col = n0 + t * 16 + l15;
#pragma unroll
    for (int r = 0; r < 8; ++r) {
      const size_t idx = (size_t)(rbase + r) * N + col;
      float val = acc[t][r];
      if constexpr (HASRES) val += resid[idx];
      if constexpr (F32OUT) outF[idx] = val;
      else                  outB[idx] = f2bf(val);
    }
  }
}

// ---------------- flash attention, sliding window 256, one wave per (b,h,16-query tile) ----
__global__ __launch_bounds__(32) void attn_kernel(const __bf16* __restrict__ q,
                                                  const __bf16* __restrict__ k,
                                                  const __bf16* __restrict__ v,
                                                  __bf16* __restrict__ ctx) {
  __shared__ __bf16 pls[16][40];   // P tile (16 queries x 32 keys), padded
  __shared__ __bf16 vt[64][40];    // V transposed: vt[hd][key], padded
  const int lane = threadIdx.x;
  const int i0 = blockIdx.x * 16;
  const int h  = blockIdx.y;
  const int b  = blockIdx.z;
  const int half = lane >> 4, l15 = lane & 15;
  const v8f zero8 = {0.f,0.f,0.f,0.f,0.f,0.f,0.f,0.f};

  // Q fragments (hd=64 -> two K=32 steps); loop invariant. Scale folded by RoPE.
  v16bf aq0, aq1;
  {
    const __bf16* qrow = q + ((size_t)(b * NS + i0 + l15)) * ND + h * HD;
    const int koff = half * 8;
#pragma unroll
    for (int e = 0; e < 8; ++e) {
      aq0[e]     = qrow[koff + e];
      aq0[8 + e] = qrow[16 + koff + e];
      aq1[e]     = qrow[32 + koff + e];
      aq1[8 + e] = qrow[48 + koff + e];
    }
  }

  v8f acc[4];
#pragma unroll
  for (int t = 0; t < 4; ++t) acc[t] = zero8;
  float mrow[8], lrow[8];
#pragma unroll
  for (int r = 0; r < 8; ++r) { mrow[r] = -1e30f; lrow[r] = 0.f; }

  for (int t = 0; t < 9; ++t) {                 // 9 x 32 keys cover [i0-256, i0+31]
    const int j0 = i0 - 256 + 32 * t;
    // prefetch next step's K/V rows (global_prefetch_b8)
    if (t < 8) {
      const int jn = min(max(j0 + 32 + lane, 0), NS - 1);
      __builtin_prefetch(k + ((size_t)(b * NS + jn)) * ND + h * HD, 0, 0);
      __builtin_prefetch(v + ((size_t)(b * NS + jn)) * ND + h * HD, 0, 0);
    }
    // ---- scores: two 16x16 tiles ----
    v8f sc[2];
#pragma unroll
    for (int s2 = 0; s2 < 2; ++s2) {
      const int jrow = j0 + s2 * 16 + l15;
      const int jc = min(max(jrow, 0), NS - 1);  // clamp; masked later
      const __bf16* krow = k + ((size_t)(b * NS + jc)) * ND + h * HD;
      const int kh = half * 8;
      v16bf bk0, bk1;
#pragma unroll
      for (int e = 0; e < 16; ++e) {
        bk0[e] = krow[kh + e + (e & 8)];
        bk1[e] = krow[32 + kh + e + (e & 8)];
      }
      v8f s = zero8;
      s = __builtin_amdgcn_wmma_f32_16x16x32_bf16(false, aq0, false, bk0, (short)0, s, false, false);
      s = __builtin_amdgcn_wmma_f32_16x16x32_bf16(false, aq1, false, bk1, (short)0, s, false, false);
      sc[s2] = s;
    }
    // ---- mask + online softmax (row reductions across 16-lane halves) ----
    const int jA = j0 + l15;
    const int jB = jA + 16;
#pragma unroll
    for (int r = 0; r < 8; ++r) {
      const int i = i0 + r + half * 8;
      const int ra = i - jA, rb = i - jB;
      const bool va = (ra >= 0) && (ra < WIN) && (jA >= 0);
      const bool vb = (rb >= 0) && (rb < WIN) && (jB >= 0);
      const float sa = va ? sc[0][r] : -1e30f;
      const float sb = vb ? sc[1][r] : -1e30f;
      float rm = fmaxf(sa, sb);
#pragma unroll
      for (int msk = 1; msk < 16; msk <<= 1) rm = fmaxf(rm, __shfl_xor(rm, msk, 32));
      const float newm  = fmaxf(mrow[r], rm);
      const float alpha = __expf(mrow[r] - newm);
      const float pa = va ? __expf(sa - newm) : 0.f;
      const float pb = vb ? __expf(sb - newm) : 0.f;
      float rs = pa + pb;
#pragma unroll
      for (int msk = 1; msk < 16; msk <<= 1) rs += __shfl_xor(rs, msk, 32);
      lrow[r] = lrow[r] * alpha + rs;
      mrow[r] = newm;
#pragma unroll
      for (int tl = 0; tl < 4; ++tl) acc[tl][r] = acc[tl][r] * alpha;
      pls[r + half * 8][l15]      = f2bf(pa);
      pls[r + half * 8][16 + l15] = f2bf(pb);
    }
    __syncthreads();
    // ---- stage V transposed: vt[hd][key 0..31] ----
    {
      const int jv = j0 + lane;
      const int jc = min(max(jv, 0), NS - 1);    // masked rows have P=0
      const __bf16* vrow = v + ((size_t)(b * NS + jc)) * ND + h * HD;
#pragma unroll
      for (int c = 0; c < 64; ++c) vt[c][lane] = vrow[c];
    }
    __syncthreads();
    // ---- P fragment (A operand 16x32) ----
    v16bf ap;
    {
      const int koff = half * 8;
#pragma unroll
      for (int e = 0; e < 8; ++e) {
        ap[e]     = pls[l15][koff + e];
        ap[8 + e] = pls[l15][16 + koff + e];
      }
    }
    // ---- O += P @ V ----
#pragma unroll
    for (int tl = 0; tl < 4; ++tl) {
      const int kh = half * 8;
      v16bf bv;
#pragma unroll
      for (int e = 0; e < 16; ++e) bv[e] = vt[tl * 16 + l15][kh + e + (e & 8)];
      acc[tl] = __builtin_amdgcn_wmma_f32_16x16x32_bf16(false, ap, false, bv,
                                                        (short)0, acc[tl], false, false);
    }
    __syncthreads();
  }
  // ---- epilogue: ctx = O / l (diag key always valid => l > 0) ----
#pragma unroll
  for (int tl = 0; tl < 4; ++tl) {
#pragma unroll
    for (int r = 0; r < 8; ++r) {
      const int row = i0 + r + half * 8;
      const int col = h * HD + tl * 16 + l15;
      ctx[((size_t)(b * NS + row)) * ND + col] = f2bf(acc[tl][r] / lrow[r]);
    }
  }
}

// ---------------- host orchestration ----------------
extern "C" void kernel_launch(void* const* d_in, const int* in_sizes, int n_in,
                              void* d_out, int out_size, void* d_ws, size_t ws_size,
                              hipStream_t stream) {
  const float* hid  = (const float*)d_in[0];
  const float* wq   = (const float*)d_in[1];
  const float* wk   = (const float*)d_in[2];
  const float* wv   = (const float*)d_in[3];
  const float* wo   = (const float*)d_in[4];
  const float* w1   = (const float*)d_in[5];
  const float* w2   = (const float*)d_in[6];
  const float* w3   = (const float*)d_in[7];
  const float* ln1g = (const float*)d_in[8];
  const float* ln1b = (const float*)d_in[9];
  const float* ln2g = (const float*)d_in[10];
  const float* ln2b = (const float*)d_in[11];
  float* out = (float*)d_out;

  const size_t DDsz = (size_t)ND * ND;      // 1M
  const size_t DIsz = (size_t)ND * NI;      // 4M
  const size_t TDsz = (size_t)NT * ND;      // 4M
  const size_t TIsz = (size_t)NT * NI;      // 16M

  char* ws = (char*)d_ws;
  auto carve = [&](size_t bytes) -> char* {
    char* p = ws;
    ws += (bytes + 255) & ~(size_t)255;
    return p;
  };
  __bf16* wq_bf  = (__bf16*)carve(DDsz * 2);
  __bf16* wk_bf  = (__bf16*)carve(DDsz * 2);
  __bf16* wv_bf  = (__bf16*)carve(DDsz * 2);
  __bf16* wo_bf  = (__bf16*)carve(DDsz * 2);
  __bf16* w1_bf  = (__bf16*)carve(DIsz * 2);
  __bf16* w3_bf  = (__bf16*)carve(DIsz * 2);
  __bf16* w2_bf  = (__bf16*)carve(DIsz * 2);
  __bf16* xln_bf = (__bf16*)carve(TDsz * 2);
  __bf16* q_bf   = (__bf16*)carve(TDsz * 2);
  __bf16* k_bf   = (__bf16*)carve(TDsz * 2);
  __bf16* v_bf   = (__bf16*)carve(TDsz * 2);
  __bf16* ctx_bf = (__bf16*)carve(TDsz * 2);
  float*  h1     = (float*) carve(TDsz * 4);
  __bf16* a1_bf  = (__bf16*)carve(TIsz * 2);
  __bf16* a3_bf  = (__bf16*)carve(TIsz * 2);

  // 1) weights -> bf16
  cvt_f32_bf16<<<2048, 256, 0, stream>>>(wq, wq_bf, DDsz);
  cvt_f32_bf16<<<2048, 256, 0, stream>>>(wk, wk_bf, DDsz);
  cvt_f32_bf16<<<2048, 256, 0, stream>>>(wv, wv_bf, DDsz);
  cvt_f32_bf16<<<2048, 256, 0, stream>>>(wo, wo_bf, DDsz);
  cvt_f32_bf16<<<4096, 256, 0, stream>>>(w1, w1_bf, DIsz);
  cvt_f32_bf16<<<4096, 256, 0, stream>>>(w3, w3_bf, DIsz);
  cvt_f32_bf16<<<4096, 256, 0, stream>>>(w2, w2_bf, DIsz);

  // 2) LN1
  ln_kernel<<<NT, 256, 0, stream>>>(hid, ln1g, ln1b, xln_bf);

  // 3) QKV projections (WMMA GEMM, bf16 out)
  dim3 gD(ND / BN, NT / BM);
  gemm_bf16<false, false><<<gD, 256, 0, stream>>>(xln_bf, wq_bf, NT, ND, ND, nullptr, q_bf, nullptr);
  gemm_bf16<false, false><<<gD, 256, 0, stream>>>(xln_bf, wk_bf, NT, ND, ND, nullptr, k_bf, nullptr);
  gemm_bf16<false, false><<<gD, 256, 0, stream>>>(xln_bf, wv_bf, NT, ND, ND, nullptr, v_bf, nullptr);

  // 4) RoPE (q gets 1/sqrt(64) folded in)
  const int ropeBlocks = (NT * (ND / 2) + 255) / 256;
  rope_kernel<<<ropeBlocks, 256, 0, stream>>>(q_bf, 0.125f);
  rope_kernel<<<ropeBlocks, 256, 0, stream>>>(k_bf, 1.0f);

  // 5) sliding-window flash attention
  attn_kernel<<<dim3(NS / 16, NH, NB), 32, 0, stream>>>(q_bf, k_bf, v_bf, ctx_bf);

  // 6) output projection + residual -> h1 (f32)
  gemm_bf16<true, true><<<gD, 256, 0, stream>>>(ctx_bf, wo_bf, NT, ND, ND, h1, nullptr, hid);

  // 7) LN2
  ln_kernel<<<NT, 256, 0, stream>>>(h1, ln2g, ln2b, xln_bf);

  // 8) MLP up projections
  dim3 gI(NI / BN, NT / BM);
  gemm_bf16<false, false><<<gI, 256, 0, stream>>>(xln_bf, w1_bf, NT, NI, ND, nullptr, a1_bf, nullptr);
  gemm_bf16<false, false><<<gI, 256, 0, stream>>>(xln_bf, w3_bf, NT, NI, ND, nullptr, a3_bf, nullptr);

  // 9) silu(a1) * a3 (in place into a1)
  silu_mul_kernel<<<8192, 256, 0, stream>>>(a1_bf, a3_bf, TIsz);

  // 10) down projection + residual -> output (f32)
  gemm_bf16<true, true><<<gD, 256, 0, stream>>>(a1_bf, w2_bf, NT, ND, NI, out, nullptr, h1);
}
